// DepthDecoder_MSF_GS_FiTAlter_DST_65764539236589
// MI455X (gfx1250) — matrix operand order repaired and checked by reading, not compiled, not run on hardware
//
#include <hip/hip_runtime.h>
#include <hip/hip_bf16.h>

// ---------------------------------------------------------------------------
// CDNA5 (gfx1250) depth-decoder: every conv is an implicit GEMM on
// v_wmma_f32_16x16x32_bf16. Activations/weights in bf16 (halves HBM traffic
// @ 23.3 TB/s, fast WMMA path), f32 accumulate.
//
// Block = 4 waves = 64(Cout) x 64(pixel) tile. K is processed in 64-wide
// double-chunks: each wave gathers TWO im2col B fragments (its 16-pixel
// sub-tile) into LDS per barrier pair, then all 4 waves consume all 8
// fragments -> 4x less global gather, 8 WMMA per barrier pair per wave.
// ---------------------------------------------------------------------------

typedef __bf16 bf16;
typedef __attribute__((ext_vector_type(16))) bf16  v16bf;
typedef __attribute__((ext_vector_type(8)))  bf16  v8bf;
typedef __attribute__((ext_vector_type(8)))  float v8f;

__device__ __forceinline__ int reflect1(int v, int n) {
  // reflect pad=1: -1 -> 1, n -> n-2
  if (v < 0)  v = -v;
  if (v >= n) v = 2 * n - 2 - v;
  return v;
}

// im2col gather of one 32-K B sub-fragment (this lane's 16 halves).
template <int KS>
__device__ __forceinline__ void gatherB(
    const bf16* __restrict__ actn, int kc, int kbaseB, int Ktrue,
    int H, int Wd, int y, int x, int ym1, int yp1, int xm1, int xp1,
    v8bf& lo, v8bf& hi) {
#pragma unroll
  for (int h = 0; h < 16; ++h) {
    int  k = kc + kbaseB + h;
    bf16 v = (bf16)0.0f;
    if (k < Ktrue) {
      int ci, yy, xx;
      if (KS == 1) {
        ci = k; yy = y; xx = x;
      } else {
        ci      = k / 9;
        int tap = k - ci * 9;
        int dy  = tap / 3;
        int dx  = tap - dy * 3;
        yy = (dy == 0) ? ym1 : ((dy == 1) ? y : yp1);
        xx = (dx == 0) ? xm1 : ((dx == 1) ? x : xp1);
      }
      v = actn[((size_t)ci * H + yy) * Wd + xx];
    }
    if (h < 8) lo[h] = v; else hi[h - 8] = v;
  }
}

// ---------------------------------------------------------------------------
// KS: 1 or 3.  ACT: 0 = ELU -> bf16 out, 1 = sigmoid -> f32 out.
//   A = packed weights [Co64][Kp] bf16 (Co64 mult of 64, Kp mult of 64, 0-pad)
//   B = im2col gather of bf16 activations (reflect pad folded into indices)
// ---------------------------------------------------------------------------
template <int KS, int ACT>
__global__ __launch_bounds__(128) void conv_wmma_bf16(
    const bf16* __restrict__ act, const bf16* __restrict__ Wp,
    const float* __restrict__ bias, void* __restrict__ outp,
    int Bn, int Cin, int H, int Wd, int Cout, int Kp) {
  // [chunk][nsub][lane][24]: 48B lane stride (16B aligned, kills bank clash)
  __shared__ bf16 ldsB[2][4][32][24];

  const int tid   = threadIdx.x;
  const int lane  = tid & 31;
  const int wave  = tid >> 5;       // 0..3: Cout sub-tile AND B gather owner
  const int ptile = blockIdx.x;     // 64-pixel tile (GEMM N)
  const int ctile = blockIdx.y;     // 64-cout tile  (GEMM M)
  const int HW    = H * Wd;
  const int Ntot  = Bn * HW;
  const int Ktrue = Cin * KS * KS;

  // Pixel this lane gathers for (B fragment nsub == wave).
  int  p      = ptile * 64 + wave * 16 + (lane & 15);
  int  pc     = (p < Ntot) ? p : (Ntot - 1);
  int  nimg   = pc / HW;
  int  rem    = pc - nimg * HW;
  int  y      = rem / Wd;
  int  x      = rem - y * Wd;

  int ym1 = y, yp1 = y, xm1 = x, xp1 = x;
  if (KS == 3) {
    ym1 = reflect1(y - 1, H);  yp1 = reflect1(y + 1, H);
    xm1 = reflect1(x - 1, Wd); xp1 = reflect1(x + 1, Wd);
  }

  const bf16* actn = act + (size_t)nimg * Cin * HW;

  const int   m      = lane & 15;   // A row within 16-row wave tile
  const int   hi     = lane >> 4;   // lane half: 0 or 1
  const int   kbaseA = hi * 16;     // A: lanes0-15 K0..15, lanes16-31 K16..31
  const int   kbaseB = hi * 16;     // B: lanes0-15 K0..15, lanes16-31 K16..31
  const bf16* Arow   = Wp + (size_t)(ctile * 64 + wave * 16 + m) * Kp;

  v8f acc[4] = {};  // 4 N-subtiles, f32

  for (int kc = 0; kc < Kp; kc += 64) {
    // speculative prefetch of next weight double-chunk -> global_prefetch_b8
    __builtin_prefetch(Arow + kc + 64, 0, 1);

    // --- two A fragments (this wave's 16 Cout rows): aligned 16B loads ---
    v8bf  a0lo = *(const v8bf*)(Arow + kc      + kbaseA);
    v8bf  a0hi = *(const v8bf*)(Arow + kc + 8  + kbaseA);
    v8bf  a1lo = *(const v8bf*)(Arow + kc + 32 + kbaseA);
    v8bf  a1hi = *(const v8bf*)(Arow + kc + 40 + kbaseA);
    v16bf aF0, aF1;
#pragma unroll
    for (int i = 0; i < 8; ++i) {
      aF0[i] = a0lo[i]; aF0[i + 8] = a0hi[i];
      aF1[i] = a1lo[i]; aF1[i + 8] = a1hi[i];
    }

    // --- B gathers for this wave's own N-subtile (both chunks in flight) ---
    v8bf b0lo, b0hi, b1lo, b1hi;
    gatherB<KS>(actn, kc,      kbaseB, Ktrue, H, Wd, y, x, ym1, yp1, xm1, xp1, b0lo, b0hi);
    gatherB<KS>(actn, kc + 32, kbaseB, Ktrue, H, Wd, y, x, ym1, yp1, xm1, xp1, b1lo, b1hi);

    // --- stage to LDS, share across the 4 Cout waves ---
    __syncthreads();  // previous iteration's reads done
    *(v8bf*)&ldsB[0][wave][lane][0] = b0lo;
    *(v8bf*)&ldsB[0][wave][lane][8] = b0hi;
    *(v8bf*)&ldsB[1][wave][lane][0] = b1lo;
    *(v8bf*)&ldsB[1][wave][lane][8] = b1hi;
    __syncthreads();  // all 8 fragments visible

#pragma unroll
    for (int ns = 0; ns < 4; ++ns) {
      v8bf  rLo = *(const v8bf*)&ldsB[0][ns][lane][0];
      v8bf  rHi = *(const v8bf*)&ldsB[0][ns][lane][8];
      v16bf bF;
#pragma unroll
      for (int i = 0; i < 8; ++i) { bF[i] = rLo[i]; bF[i + 8] = rHi[i]; }
      acc[ns] = __builtin_amdgcn_wmma_f32_16x16x32_bf16(
          false, aF0, false, bF, (short)0, acc[ns], false, false);
    }
#pragma unroll
    for (int ns = 0; ns < 4; ++ns) {
      v8bf  rLo = *(const v8bf*)&ldsB[1][ns][lane][0];
      v8bf  rHi = *(const v8bf*)&ldsB[1][ns][lane][8];
      v16bf bF;
#pragma unroll
      for (int i = 0; i < 8; ++i) { bF[i] = rLo[i]; bF[i + 8] = rHi[i]; }
      acc[ns] = __builtin_amdgcn_wmma_f32_16x16x32_bf16(
          false, aF1, false, bF, (short)0, acc[ns], false, false);
    }
  }

  // Epilogue: bias + activation. C/D layout: M = r + 8*hi, N = lane&15.
#pragma unroll
  for (int ns = 0; ns < 4; ++ns) {
    int pp = ptile * 64 + ns * 16 + (lane & 15);
    if (pp >= Ntot) continue;
    int ni = pp / HW;
    int rr = pp - ni * HW;
    int yy = rr / Wd;
    int xx = rr - yy * Wd;
#pragma unroll
    for (int r = 0; r < 8; ++r) {
      int cout = ctile * 64 + wave * 16 + r + hi * 8;
      if (cout >= Cout) continue;
      float  v   = acc[ns][r] + bias[cout];
      size_t off = (((size_t)ni * Cout + cout) * H + yy) * Wd + xx;
      if (ACT == 0) {
        v = (v > 0.0f) ? v : (__expf(v) - 1.0f);           // ELU
        ((bf16*)outp)[off] = (bf16)v;
      } else {
        ((float*)outp)[off] = 1.0f / (1.0f + __expf(-v));  // sigmoid -> f32
      }
    }
  }
}

// ---------------------------------------------------------------------------
// Elementwise helpers
// ---------------------------------------------------------------------------
__global__ __launch_bounds__(256) void cvt_f32_to_bf16(
    const float* __restrict__ in, bf16* __restrict__ out, int n) {
  int i = blockIdx.x * blockDim.x + threadIdx.x;
  if (i < n) out[i] = (bf16)in[i];
}

// Pack (Cout,Cin,kh,kw) f32 weights -> [Co64][Kp] bf16, zero-padded.
__global__ __launch_bounds__(256) void pack_weight_bf16(
    const float* __restrict__ w, bf16* __restrict__ wp,
    int Cout, int Ktrue, int Kp, int n) {
  int i = blockIdx.x * blockDim.x + threadIdx.x;
  if (i >= n) return;
  int   co = i / Kp;
  int   k  = i - co * Kp;
  float v  = (co < Cout && k < Ktrue) ? w[(size_t)co * Ktrue + k] : 0.0f;
  wp[i] = (bf16)v;
}

// out = base + up_{2^s1}(t1) + up_{2^s2}(t2) + up_{2^s3}(t3)  (bf16 in/out,
// f32 accumulate). Null pointers skip a term; null base means 0.
__global__ __launch_bounds__(256) void fuse_upsample_add(
    bf16* __restrict__ out, const bf16* __restrict__ base,
    const bf16* __restrict__ t1, int s1,
    const bf16* __restrict__ t2, int s2,
    const bf16* __restrict__ t3, int s3,
    int H, int W, int n) {
  int i = blockIdx.x * blockDim.x + threadIdx.x;
  if (i >= n) return;
  int x  = i % W;
  int t  = i / W;
  int y  = t % H;
  int nc = t / H;  // fused (batch*channel) index — identical for all operands
  float acc = base ? (float)base[i] : 0.0f;
  if (t1) acc += (float)t1[((size_t)nc * (H >> s1) + (y >> s1)) * (W >> s1) + (x >> s1)];
  if (t2) acc += (float)t2[((size_t)nc * (H >> s2) + (y >> s2)) * (W >> s2) + (x >> s2)];
  if (t3) acc += (float)t3[((size_t)nc * (H >> s3) + (y >> s3)) * (W >> s3) + (x >> s3)];
  out[i] = (bf16)acc;
}

// ---------------------------------------------------------------------------
// Host orchestration
// ---------------------------------------------------------------------------
struct CSpec { int co, ci, ks; };
static const CSpec SPECS[22] = {
    {512, 512, 3},  // 0  pc0_4
    {256, 512, 1},  // 1  c0_43
    {128, 512, 1},  // 2  c0_42
    { 64, 512, 1},  // 3  c0_41
    {128, 256, 1},  // 4  c0_32
    { 64, 256, 1},  // 5  c0_31
    { 64, 128, 1},  // 6  c0_21
    {256, 256, 3},  // 7  pc1_3
    {128, 128, 3},  // 8  pc1_2
    { 64,  64, 3},  // 9  pc1_1
    {128, 256, 1},  // 10 c1_32
    { 64, 256, 1},  // 11 c1_31
    { 64, 128, 1},  // 12 c1_21
    {128, 128, 3},  // 13 pc2_2
    { 64,  64, 3},  // 14 pc2_1
    { 64, 128, 1},  // 15 c2_21
    { 64,  64, 3},  // 16 pc3_1
    { 64,  64, 3},  // 17 pc3_0
    { 64,  64, 1},  // 18 c3_10
    { 32,  64, 3},  // 19 pc4_0
    { 16,  32, 3},  // 20 pc5_0
    {  1,  16, 3},  // 21 disp
};

static inline int cdiv(int a, int b) { return (a + b - 1) / b; }

extern "C" void kernel_launch(void* const* d_in, const int* in_sizes, int n_in,
                              void* d_out, int out_size, void* d_ws, size_t ws_size,
                              hipStream_t stream) {
  (void)in_sizes; (void)n_in; (void)out_size; (void)ws_size;
  const int B = 4;

  // Bump allocator over workspace (256B aligned regions).
  char*  wsb = (char*)d_ws;
  size_t off = 0;
  auto alloc = [&](size_t elems) -> bf16* {
    bf16* p = (bf16*)(wsb + off);
    off += (elems * sizeof(bf16) + 255) & ~(size_t)255;
    return p;
  };

  // --- Pack all 22 weight tensors to bf16 [Co64][Kp] (deterministic each call)
  bf16* wp[22];
  for (int i = 0; i < 22; ++i) {
    int K    = SPECS[i].ci * SPECS[i].ks * SPECS[i].ks;
    int Kp   = (K + 63) & ~63;
    int Co64 = (SPECS[i].co + 63) & ~63;
    int n    = Co64 * Kp;
    wp[i]    = alloc((size_t)n);
    pack_weight_bf16<<<cdiv(n, 256), 256, 0, stream>>>(
        (const float*)d_in[7 + 2 * i], wp[i], SPECS[i].co, K, Kp, n);
  }

  // --- bf16 copies of encoder features
  auto cvt = [&](const float* src, size_t n) -> bf16* {
    bf16* dst = alloc(n);
    cvt_f32_to_bf16<<<cdiv((int)n, 256), 256, 0, stream>>>(src, dst, (int)n);
    return dst;
  };
  bf16* f0b = cvt((const float*)d_in[0], (size_t)B * 64  * 192 * 640);
  bf16* f1b = cvt((const float*)d_in[1], (size_t)B * 64  * 96  * 320);
  bf16* f2b = cvt((const float*)d_in[2], (size_t)B * 128 * 48  * 160);
  bf16* f3b = cvt((const float*)d_in[3], (size_t)B * 256 * 24  * 80);
  bf16* f4b = cvt((const float*)d_in[4], (size_t)B * 512 * 12  * 40);

  auto conv = [&](int idx, const bf16* in, void* out, int H, int W, int act) {
    const CSpec s = SPECS[idx];
    int K    = s.ci * s.ks * s.ks;
    int Kp   = (K + 63) & ~63;
    int Co64 = (s.co + 63) & ~63;
    dim3 grid(cdiv(B * H * W, 64), Co64 / 64);
    const float* bias = (const float*)d_in[8 + 2 * idx];
    if (s.ks == 3) {
      if (act == 0)
        conv_wmma_bf16<3, 0><<<grid, 128, 0, stream>>>(in, wp[idx], bias, out, B, s.ci, H, W, s.co, Kp);
      else
        conv_wmma_bf16<3, 1><<<grid, 128, 0, stream>>>(in, wp[idx], bias, out, B, s.ci, H, W, s.co, Kp);
    } else {
      conv_wmma_bf16<1, 0><<<grid, 128, 0, stream>>>(in, wp[idx], bias, out, B, s.ci, H, W, s.co, Kp);
    }
  };
  auto fuse = [&](bf16* out, const bf16* base, int C, int H, int W,
                  const bf16* t1, int s1, const bf16* t2, int s2,
                  const bf16* t3, int s3) {
    int n = B * C * H * W;
    fuse_upsample_add<<<cdiv(n, 256), 256, 0, stream>>>(out, base, t1, s1, t2, s2, t3, s3, H, W, n);
  };

  // --- Intermediates (bf16, NCHW)
  bf16* d0_4  = alloc((size_t)B * 512 * 12 * 40);
  bf16* t43   = alloc((size_t)B * 256 * 12 * 40);
  bf16* t42   = alloc((size_t)B * 128 * 12 * 40);
  bf16* t41   = alloc((size_t)B * 64  * 12 * 40);
  bf16* t32   = alloc((size_t)B * 128 * 24 * 80);
  bf16* t31   = alloc((size_t)B * 64  * 24 * 80);
  bf16* t21   = alloc((size_t)B * 64  * 48 * 160);
  bf16* sum1  = alloc((size_t)B * 256 * 24 * 80);
  bf16* d1_3  = alloc((size_t)B * 256 * 24 * 80);
  bf16* sum2  = alloc((size_t)B * 128 * 48 * 160);
  bf16* d1_2  = alloc((size_t)B * 128 * 48 * 160);
  bf16* sum3  = alloc((size_t)B * 64  * 96 * 320);
  bf16* d1_1  = alloc((size_t)B * 64  * 96 * 320);
  bf16* t132  = alloc((size_t)B * 128 * 24 * 80);
  bf16* t131  = alloc((size_t)B * 64  * 24 * 80);
  bf16* t121  = alloc((size_t)B * 64  * 48 * 160);
  bf16* sum22 = alloc((size_t)B * 128 * 48 * 160);
  bf16* d2_2  = alloc((size_t)B * 128 * 48 * 160);
  bf16* sum21 = alloc((size_t)B * 64  * 96 * 320);
  bf16* d2_1  = alloc((size_t)B * 64  * 96 * 320);
  bf16* t221  = alloc((size_t)B * 64  * 48 * 160);
  bf16* sum31 = alloc((size_t)B * 64  * 96 * 320);
  bf16* d3_1  = alloc((size_t)B * 64  * 96 * 320);
  bf16* t310  = alloc((size_t)B * 64  * 96 * 320);
  bf16* sum30 = alloc((size_t)B * 64  * 192 * 640);
  bf16* d3_0  = alloc((size_t)B * 64  * 192 * 640);
  bf16* x4    = alloc((size_t)B * 32  * 192 * 640);
  bf16* xu    = alloc((size_t)B * 32  * 384 * 1280);
  bf16* x5    = alloc((size_t)B * 16  * 384 * 1280);

  // --- Decoder pipeline (stream-ordered) ---
  conv(0, f4b, d0_4, 12, 40, 0);                                   // pc0_4
  conv(1, d0_4, t43, 12, 40, 0);                                   // c0_43
  conv(2, d0_4, t42, 12, 40, 0);                                   // c0_42
  conv(3, d0_4, t41, 12, 40, 0);                                   // c0_41
  conv(4, f3b,  t32, 24, 80, 0);                                   // c0_32
  conv(5, f3b,  t31, 24, 80, 0);                                   // c0_31
  conv(6, f2b,  t21, 48, 160, 0);                                  // c0_21

  fuse(sum1, f3b, 256, 24, 80,  t43, 1, nullptr, 0, nullptr, 0);
  conv(7, sum1, d1_3, 24, 80, 0);                                  // pc1_3

  fuse(sum2, f2b, 128, 48, 160, t32, 1, t42, 2, nullptr, 0);
  conv(8, sum2, d1_2, 48, 160, 0);                                 // pc1_2

  fuse(sum3, f1b, 64, 96, 320,  t21, 1, t31, 2, t41, 3);
  conv(9, sum3, d1_1, 96, 320, 0);                                 // pc1_1

  conv(10, d1_3, t132, 24, 80, 0);                                 // c1_32
  conv(11, d1_3, t131, 24, 80, 0);                                 // c1_31
  conv(12, d1_2, t121, 48, 160, 0);                                // c1_21

  fuse(sum22, d1_2, 128, 48, 160, t132, 1, nullptr, 0, nullptr, 0);
  conv(13, sum22, d2_2, 48, 160, 0);                               // pc2_2

  fuse(sum21, d1_1, 64, 96, 320,  t121, 1, t131, 2, nullptr, 0);
  conv(14, sum21, d2_1, 96, 320, 0);                               // pc2_1

  conv(15, d2_2, t221, 48, 160, 0);                                // c2_21
  fuse(sum31, d2_1, 64, 96, 320,  t221, 1, nullptr, 0, nullptr, 0);
  conv(16, sum31, d3_1, 96, 320, 0);                               // pc3_1

  conv(18, d3_1, t310, 96, 320, 0);                                // c3_10
  fuse(sum30, f0b, 64, 192, 640,  t310, 1, nullptr, 0, nullptr, 0);
  conv(17, sum30, d3_0, 192, 640, 0);                              // pc3_0

  conv(19, d3_0, x4, 192, 640, 0);                                 // pc4_0
  fuse(xu, nullptr, 32, 384, 1280, x4, 1, nullptr, 0, nullptr, 0); // up x2
  conv(20, xu, x5, 384, 1280, 0);                                  // pc5_0
  conv(21, x5, d_out, 384, 1280, 1);                               // disp (sigmoid, f32)
}
